// ECGTransformer_44676249813048
// MI455X (gfx1250) — matrix operand order repaired and checked by reading, not compiled, not run on hardware
//
#include <hip/hip_runtime.h>
#include <hip/hip_bf16.h>
#include <math.h>

typedef __attribute__((ext_vector_type(16))) _Float16 v16h;
typedef __attribute__((ext_vector_type(8)))  _Float16 v8h;
typedef __attribute__((ext_vector_type(8)))  float    v8f;

#define B_   2
#define L_   15000
#define D_   96
#define H_   2
#define HD_  48
#define NL_  4
#define NC_  4
#define W_   256
#define S_   15360
#define FF_  384
#define M_   (B_*S_)       // 30720 rows
#define BH_  (B_*H_)       // 4
#define CH_  (S_/W_)       // 60 chunks
#define NEGV (-1.0e9f)
#define BN_SCALE 0.9999950000374997f   // 1/sqrt(1+1e-5)
#define QSCALE   0.14433756729740643f  // 1/sqrt(48)

// ---------------- wave reductions (wave32) ----------------
__device__ __forceinline__ float wred_sum(float v) {
#pragma unroll
  for (int o = 16; o > 0; o >>= 1) v += __shfl_xor(v, o, 32);
  return v;
}
__device__ __forceinline__ float wred_max(float v) {
#pragma unroll
  for (int o = 16; o > 0; o >>= 1) v = fmaxf(v, __shfl_xor(v, o, 32));
  return v;
}

// ---------------- WMMA GEMM: out = act(A[M,K]*Bpack[K,N] + bias) ------------
// A: f16 row-major. Bpack: fragment-packed f16 (see pack_b). One wave computes
// a 16x96 strip (6 accumulators) so each A fragment feeds 6 WMMAs. N%96==0.
__global__ __launch_bounds__(64)
void gemm_wmma(const _Float16* __restrict__ A, const _Float16* __restrict__ Bp,
               const float* __restrict__ bias, float* __restrict__ o32,
               _Float16* __restrict__ o16, int M, int K, int N, int act)
{
  const int nblk  = N / 96;                  // N-strips of 6 tiles
  const int total = (M >> 4) * nblk;
  int tile = blockIdx.x * 2 + threadIdx.y;
  if (tile >= total) return;                 // wave-uniform
  int mt = tile / nblk, nb = tile - mt * nblk;
  int lane = threadIdx.x, g = lane >> 4, r = lane & 15;
  const int ntiles = N >> 4;
  v8f acc[6];
#pragma unroll
  for (int j = 0; j < 6; ++j) acc[j] = (v8f){0.f,0.f,0.f,0.f,0.f,0.f,0.f,0.f};
  const int ks = K >> 5;
  const _Float16* arow = A + (size_t)(mt * 16 + r) * K + 16 * g;
  const _Float16* bptr = Bp + ((size_t)(nb * 6) * 32 + lane) * 16;
  for (int kt = 0; kt < ks; ++kt) {
    union { v16h v; v8h p[2]; } a;
    a.p[0] = *(const v8h*)(arow);
    a.p[1] = *(const v8h*)(arow + 8);
#pragma unroll
    for (int j = 0; j < 6; ++j) {
      v16h b = *(const v16h*)(bptr + j * 512);
      acc[j] = __builtin_amdgcn_wmma_f32_16x16x32_f16(false, a.v, false, b,
                                                      (short)0, acc[j], false, false);
    }
    arow += 32;
    bptr += (size_t)ntiles * 512;
  }
#pragma unroll
  for (int j = 0; j < 6; ++j) {
    int col = (nb * 6 + j) * 16 + r;
    float bs = bias ? bias[col] : 0.f;
#pragma unroll
    for (int e = 0; e < 8; ++e) {
      int row = mt * 16 + e + 8 * g;
      float y = acc[j][e] + bs;
      if (act == 1)      y = fmaxf(y, 0.f);
      else if (act == 2) y = 0.5f * y * (1.f + erff(y * 0.70710678118654752f));
      size_t idx = (size_t)row * N + col;
      if (o32) o32[idx] = y;
      if (o16) o16[idx] = (_Float16)y;
    }
  }
}

// ---------------- pack f32 weight [K,N] into WMMA B-fragment f16 layout -----
__global__ void pack_b(const float* __restrict__ Wsrc, _Float16* __restrict__ dst,
                       int K, int N)
{
  int idx = blockIdx.x * 256 + threadIdx.x;
  if (idx >= K * N) return;
  int h = idx & 15, lane = (idx >> 4) & 31, t = idx >> 9;
  int ntiles = N >> 4;
  int nt = t % ntiles, kt = t / ntiles;
  int kk = kt * 32 + ((lane >> 4) << 4) + h;   // K row
  int nn = nt * 16 + (lane & 15);              // N col
  dst[idx] = (_Float16)Wsrc[(size_t)kk * N + nn];
}

// ---------------- conv1d (+optional BN eval + relu + accumulate) ------------
__global__ void conv1d_k(const float* __restrict__ in, const float* __restrict__ w,
                         const float* __restrict__ bias, const float* __restrict__ bng,
                         const float* __restrict__ bnb, float* __restrict__ out,
                         int Cin, int Cout, int Kw, int pad, int outCoff,
                         int outCtot, int relu, int accum)
{
  int idx = blockIdx.x * 256 + threadIdx.x;
  if (idx >= B_ * Cout * L_) return;
  int l  = idx % L_;
  int oc = (idx / L_) % Cout;
  int bb = idx / (L_ * Cout);
  float acc = bias[oc];
  for (int ci = 0; ci < Cin; ++ci) {
    const float* ip = in + ((size_t)bb * Cin + ci) * L_;
    const float* wp = w + ((size_t)oc * Cin + ci) * Kw;
    for (int k = 0; k < Kw; ++k) {
      int p = l + k - pad;
      if (p >= 0 && p < L_) acc = fmaf(ip[p], wp[k], acc);
    }
  }
  if (bng) acc = acc * (bng[oc] * BN_SCALE) + bnb[oc];
  if (relu) acc = fmaxf(acc, 0.f);
  float* op = out + ((size_t)bb * outCtot + outCoff + oc) * L_ + l;
  if (accum) *op += acc; else *op = acc;
}

__global__ void bnrelu_k(float* __restrict__ x, const float* __restrict__ g,
                         const float* __restrict__ b)
{
  int idx = blockIdx.x * 256 + threadIdx.x;
  if (idx >= B_ * D_ * L_) return;
  int c = (idx / L_) % D_;
  x[idx] = fmaxf(x[idx] * (g[c] * BN_SCALE) + b[c], 0.f);
}

// -------- LN over D=96, wave-per-row: out = LN(X+R)*g+b; optional += --------
__global__ __launch_bounds__(256)
void ln_kernel(const float* __restrict__ X, const float* __restrict__ R,
               const float* __restrict__ gam, const float* __restrict__ bet,
               float* __restrict__ o32, _Float16* __restrict__ o16,
               float* __restrict__ addto)
{
  int row = blockIdx.x * 8 + threadIdx.y;
  if (row >= M_) return;
  int lane = threadIdx.x;
  size_t base = (size_t)row * D_;
  float x[3];
#pragma unroll
  for (int j = 0; j < 3; ++j) {
    int c = lane + 32 * j;
    x[j] = X[base + c] + (R ? R[base + c] : 0.f);
  }
  float m = wred_sum(x[0] + x[1] + x[2]) * (1.f / 96.f);
  float d0 = x[0] - m, d1 = x[1] - m, d2 = x[2] - m;
  float var = wred_sum(d0 * d0 + d1 * d1 + d2 * d2) * (1.f / 96.f);
  float inv = rsqrtf(var + 1e-5f);
  float dd[3] = {d0, d1, d2};
#pragma unroll
  for (int j = 0; j < 3; ++j) {
    int c = lane + 32 * j;
    float y = dd[j] * inv * gam[c] + bet[c];
    if (o32)   o32[base + c] = y;
    if (o16)   o16[base + c] = (_Float16)y;
    if (addto) addto[base + c] += y;
  }
}

// -------- fused: hid=LN(h,n1); emb=LN(hid+pos+tok, emb_ln) -> f32 + f16 -----
__global__ __launch_bounds__(256)
void ln_emb_k(const float* __restrict__ Hb, const float* __restrict__ n1g,
              const float* __restrict__ n1b, const float* __restrict__ pos,
              const float* __restrict__ tok, const float* __restrict__ eg,
              const float* __restrict__ eb, float* __restrict__ e32,
              _Float16* __restrict__ e16)
{
  int row = blockIdx.x * 8 + threadIdx.y;
  if (row >= M_) return;
  int lane = threadIdx.x;
  int s = row % S_;
  size_t base = (size_t)row * D_;
  float x[3];
#pragma unroll
  for (int j = 0; j < 3; ++j) x[j] = Hb[base + lane + 32 * j];
  float m = wred_sum(x[0] + x[1] + x[2]) * (1.f / 96.f);
  float d0 = x[0] - m, d1 = x[1] - m, d2 = x[2] - m;
  float var = wred_sum(d0 * d0 + d1 * d1 + d2 * d2) * (1.f / 96.f);
  float inv = rsqrtf(var + 1e-5f);
  float t[3];
#pragma unroll
  for (int j = 0; j < 3; ++j) {
    int c = lane + 32 * j;
    float hid = (x[j] - m) * inv * n1g[c] + n1b[c];
    t[j] = hid + pos[(size_t)s * D_ + c] + tok[c];
  }
  float m2 = wred_sum(t[0] + t[1] + t[2]) * (1.f / 96.f);
  float e0 = t[0] - m2, e1 = t[1] - m2, e2 = t[2] - m2;
  float v2 = wred_sum(e0 * e0 + e1 * e1 + e2 * e2) * (1.f / 96.f);
  float i2 = rsqrtf(v2 + 1e-5f);
  float ee[3] = {e0, e1, e2};
#pragma unroll
  for (int j = 0; j < 3; ++j) {
    int c = lane + 32 * j;
    float y = ee[j] * i2 * eg[c] + eb[c];
    e32[base + c] = y;
    e16[base + c] = (_Float16)y;
  }
}

// -------- frontend: transpose [B,96,L]->rows, LN, +sincos PE, pad to S ------
__global__ __launch_bounds__(256)
void ln_pe_pad_k(const float* __restrict__ cat, const float* __restrict__ g,
                 const float* __restrict__ b, float* __restrict__ Ho)
{
  int row = blockIdx.x * 8 + threadIdx.y;
  if (row >= M_) return;
  int lane = threadIdx.x;
  int s = row % S_, bb = row / S_;
  size_t base = (size_t)row * D_;
  if (s >= L_) {
#pragma unroll
    for (int j = 0; j < 3; ++j) Ho[base + lane + 32 * j] = 0.f;
    return;
  }
  float x[3];
#pragma unroll
  for (int j = 0; j < 3; ++j) {
    int c = lane + 32 * j;
    x[j] = cat[((size_t)bb * D_ + c) * L_ + s];
  }
  float m = wred_sum(x[0] + x[1] + x[2]) * (1.f / 96.f);
  float d0 = x[0] - m, d1 = x[1] - m, d2 = x[2] - m;
  float var = wred_sum(d0 * d0 + d1 * d1 + d2 * d2) * (1.f / 96.f);
  float inv = rsqrtf(var + 1e-5f);
  float dd[3] = {d0, d1, d2};
#pragma unroll
  for (int j = 0; j < 3; ++j) {
    int c = lane + 32 * j;
    float y = dd[j] * inv * g[c] + b[c];
    int ii = c & ~1;
    float div = expf(-(float)ii * (logf(10000.f) / (float)D_));
    float ang = (float)s * div;
    y += (c & 1) ? cosf(ang) : sinf(ang);
    Ho[base + c] = y;
  }
}

// -------- repack [B,S,96] f32 -> per-head f16, HD padded 48->64 -------------
__global__ void repack_head_k(const float* __restrict__ X, _Float16* __restrict__ out,
                              int transpose, float scale)
{
  int idx = blockIdx.x * 256 + threadIdx.x;
  if (idx >= BH_ * S_ * 64) return;
  int hd = idx & 63;
  int s  = (idx >> 6) % S_;
  int bh = idx / (S_ * 64);
  int b0 = bh / H_, hh = bh % H_;
  float v = 0.f;
  if (hd < HD_) v = X[((size_t)b0 * S_ + s) * D_ + hh * HD_ + hd] * scale;
  if (transpose) out[((size_t)bh * 64 + hd) * S_ + s] = (_Float16)v;
  else           out[idx] = (_Float16)v;
}

// -------- sliding-window attention: one block = (bh, chunk, 16-query strip) -
__global__ __launch_bounds__(256)
void band_attn_k(const _Float16* __restrict__ qh, const _Float16* __restrict__ kh,
                 const _Float16* __restrict__ vT, _Float16* __restrict__ att16)
{
  __shared__ __attribute__((aligned(32))) float smem[16 * 768];   // 48 KB scores
  _Float16* pf = (_Float16*)smem;                                 // aliased P(f16)
  int bh = blockIdx.y;
  int chunk = blockIdx.x >> 4, strip = blockIdx.x & 15;
  int q0 = chunk * W_ + strip * 16;      // absolute query start
  int kbase = chunk * W_ - W_;           // absolute key of slot 0 (may be <0)
  int wv = threadIdx.y, lane = threadIdx.x;
  int g = lane >> 4, r = lane & 15;
  int b0 = bh / H_, hh = bh % H_;
  const _Float16* qbase = qh + ((size_t)bh * S_ + q0) * 64;

  // A fragments (Q strip, K-dim=64) are invariant across all 48 key tiles.
  union { v16h v; v8h p[2]; } a0, a1;
  {
    const _Float16* ap = qbase + (size_t)r * 64 + 16 * g;
    a0.p[0] = *(const v8h*)(ap);
    a0.p[1] = *(const v8h*)(ap + 8);
    a1.p[0] = *(const v8h*)(ap + 32);
    a1.p[1] = *(const v8h*)(ap + 40);
  }

  // phase 1: scores = Q*K^T (+band/kmask), 48 N-tiles over 8 waves
  for (int nt = wv; nt < 48; nt += 8) {
    v8f acc = {0.f,0.f,0.f,0.f,0.f,0.f,0.f,0.f};
    int key = kbase + nt * 16 + r;
#pragma unroll
    for (int kt = 0; kt < 2; ++kt) {
      union { v16h v; _Float16 e[16]; } bu;
      if (key >= 0 && key < S_) {
        bu.v = *(const v16h*)(kh + ((size_t)bh * S_ + key) * 64 + kt * 32 + 16 * g);
      } else {
#pragma unroll
        for (int t = 0; t < 16; ++t) bu.e[t] = (_Float16)0.f;
      }
      acc = __builtin_amdgcn_wmma_f32_16x16x32_f16(false, kt ? a1.v : a0.v, false,
                                                   bu.v, (short)0, acc, false, false);
    }
    bool valid = (key >= 0) && (key < L_);   // kmask: in-sequence & unpadded
#pragma unroll
    for (int e = 0; e < 8; ++e) {
      int qm = e + 8 * g;
      float sv = acc[e];
      int rel = key - (q0 + qm);
      if (rel < -W_ || rel > W_) sv = NEGV;
      else if (!valid) sv -= 10000.f;
      smem[qm * 768 + nt * 16 + r] = sv;
    }
  }
  __syncthreads();

  // phase 2a: softmax stats, all score reads complete before P overwrites
  float vals[2][24]; float pinv[2];
#pragma unroll
  for (int rr = 0; rr < 2; ++rr) {
    int row = wv * 2 + rr;
    float mx = -3e38f;
#pragma unroll
    for (int j = 0; j < 24; ++j) {
      float v = smem[row * 768 + lane + 32 * j];
      vals[rr][j] = v; mx = fmaxf(mx, v);
    }
    mx = wred_max(mx);
    float sum = 0.f;
#pragma unroll
    for (int j = 0; j < 24; ++j) {
      float e = __expf(vals[rr][j] - mx);
      vals[rr][j] = e; sum += e;
    }
    sum = wred_sum(sum);
    pinv[rr] = 1.f / sum;
  }
  __syncthreads();
  // phase 2b: write P as f16 (aliases score buffer)
#pragma unroll
  for (int rr = 0; rr < 2; ++rr) {
    int row = wv * 2 + rr;
#pragma unroll
    for (int j = 0; j < 24; ++j)
      pf[row * 768 + lane + 32 * j] = (_Float16)(vals[rr][j] * pinv[rr]);
  }
  __syncthreads();

  // phase 3: out = P * V. 6 waves: (K-half 0/1) x (hd tile 0..2), K-depth 384 each.
  v8f acc = {0.f,0.f,0.f,0.f,0.f,0.f,0.f,0.f};
  int gid = wv / 3, nt3 = wv % 3;
  if (wv < 6) {
    const _Float16* vrow = vT + ((size_t)bh * 64 + nt3 * 16 + r) * S_;
    for (int kt = gid * 12; kt < gid * 12 + 12; ++kt) {
      union { v16h v; v8h p[2]; } a;
      a.p[0] = *(const v8h*)&pf[r * 768 + kt * 32 + 16 * g];
      a.p[1] = *(const v8h*)&pf[r * 768 + kt * 32 + 16 * g + 8];
      int k0 = kbase + kt * 32 + 16 * g;
      union { v16h v; _Float16 e[16]; } bu;
      if (k0 >= 0 && k0 + 16 <= S_) {
        bu.v = *(const v16h*)(vrow + k0);
      } else {
#pragma unroll
        for (int t = 0; t < 16; ++t) {
          int kk = k0 + t;
          bu.e[t] = (kk >= 0 && kk < S_) ? vrow[kk] : (_Float16)0.f;
        }
      }
      acc = __builtin_amdgcn_wmma_f32_16x16x32_f16(false, a.v, false, bu.v,
                                                   (short)0, acc, false, false);
    }
  }
  __syncthreads();   // all P reads complete; smem may now hold f32 partials
  if (wv < 6) {
#pragma unroll
    for (int e = 0; e < 8; ++e) {
      int qm = e + 8 * g;
      smem[gid * 768 + qm * 48 + nt3 * 16 + r] = acc[e];
    }
  }
  __syncthreads();
  // combine halves and store att (f16) with head interleave
  int tid = wv * 32 + lane;
  for (int idx = tid; idx < 768; idx += 256) {
    int qm = idx / 48, hd = idx - qm * 48;
    float sum = smem[idx] + smem[768 + idx];
    int qp = q0 + qm;
    att16[((size_t)b0 * S_ + qp) * D_ + hh * HD_ + hd] = (_Float16)sum;
  }
}

// ---------------- misc elementwise / head kernels ---------------------------
__global__ void add_k(float* __restrict__ a, const float* __restrict__ b, int n)
{
  int i = blockIdx.x * 256 + threadIdx.x;
  if (i < n) a[i] += b[i];
}

__global__ __launch_bounds__(256)
void pool_softmax_k(const float* __restrict__ Hb, const float* __restrict__ pw,
                    const float* __restrict__ pb, float* __restrict__ sc)
{
  __shared__ float red[256];
  __shared__ float s_mx, s_sum;
  int b0 = blockIdx.x, tid = threadIdx.x;
  float mx = -3e38f;
  for (int s = tid; s < S_; s += 256) {
    float lg;
    if (s < L_) {
      const float* hp = Hb + ((size_t)b0 * S_ + s) * D_;
      float acc = pb[0];
      for (int d = 0; d < D_; ++d) acc = fmaf(hp[d], pw[d], acc);
      lg = acc;
    } else lg = NEGV;
    sc[(size_t)b0 * S_ + s] = lg;
    mx = fmaxf(mx, lg);
  }
  red[tid] = mx; __syncthreads();
  for (int o = 128; o > 0; o >>= 1) { if (tid < o) red[tid] = fmaxf(red[tid], red[tid + o]); __syncthreads(); }
  if (tid == 0) s_mx = red[0];
  __syncthreads();
  float mxv = s_mx, sum = 0.f;
  for (int s = tid; s < S_; s += 256) sum += __expf(sc[(size_t)b0 * S_ + s] - mxv);
  red[tid] = sum; __syncthreads();
  for (int o = 128; o > 0; o >>= 1) { if (tid < o) red[tid] += red[tid + o]; __syncthreads(); }
  if (tid == 0) s_sum = red[0];
  __syncthreads();
  float invs = 1.f / s_sum;
  for (int s = tid; s < S_; s += 256) {
    size_t i = (size_t)b0 * S_ + s;
    sc[i] = __expf(sc[i] - mxv) * invs;
  }
}

__global__ __launch_bounds__(256)
void feat_k(const float* __restrict__ Hb, const float* __restrict__ sc,
            float* __restrict__ feat)
{
  __shared__ float red[256];
  int b0 = blockIdx.x / D_, d = blockIdx.x % D_, tid = threadIdx.x;
  float acc = 0.f;
  for (int s = tid; s < S_; s += 256) {
    float w = 0.5f * sc[(size_t)b0 * S_ + s] + ((s < L_) ? (0.5f / (float)L_) : 0.f);
    acc = fmaf(Hb[((size_t)b0 * S_ + s) * D_ + d], w, acc);
  }
  red[tid] = acc; __syncthreads();
  for (int o = 128; o > 0; o >>= 1) { if (tid < o) red[tid] += red[tid + o]; __syncthreads(); }
  if (tid == 0) feat[b0 * D_ + d] = red[0];
}

__global__ __launch_bounds__(256)
void head_k(const float* __restrict__ feat, const float* __restrict__ fc1w,
            const float* __restrict__ fc1b, const float* __restrict__ fc2w,
            const float* __restrict__ fc2b, float* __restrict__ out)
{
  __shared__ float z[B_ * D_];
  int tid = threadIdx.x;
  if (tid < B_ * D_) {
    int b0 = tid / D_, d = tid % D_;
    float a = fc1b[d];
    for (int k = 0; k < D_; ++k) a = fmaf(feat[b0 * D_ + k], fc1w[(size_t)k * D_ + d], a);
    z[tid] = fmaxf(a, 0.f) + feat[tid];
  }
  __syncthreads();
  if (tid < B_ * NC_) {
    int b0 = tid / NC_, c = tid % NC_;
    float o = fc2b[c];
    for (int k = 0; k < D_; ++k) o = fmaf(z[b0 * D_ + k], fc2w[(size_t)k * NC_ + c], o);
    out[tid] = o;
  }
}

// ---------------- host orchestration ---------------------------------------
extern "C" void kernel_launch(void* const* d_in, const int* in_sizes, int n_in,
                              void* d_out, int out_size, void* d_ws, size_t ws_size,
                              hipStream_t stream)
{
  (void)in_sizes; (void)n_in; (void)out_size; (void)ws_size;
#define P(i) ((const float*)d_in[i])
  const float* x = P(0);   // [B,L,1] == [B,1,L]; d_in[1] = mask (all True; s<L)

  // ---- workspace carve (256B aligned) ----
  char* ws = (char*)d_ws;
  size_t off = 0;
  auto alloc = [&](size_t bytes) -> char* {
    char* p = ws + off; off += (bytes + 255) & ~(size_t)255; return p;
  };
  float*    H32   = (float*)alloc((size_t)M_ * D_ * 4);
  float*    EMB32 = (float*)alloc((size_t)M_ * D_ * 4);
  float*    R32   = (float*)alloc((size_t)M_ * D_ * 4);
  float*    Y96   = (float*)alloc((size_t)M_ * D_ * 4);
  _Float16* X16   = (_Float16*)alloc((size_t)M_ * D_ * 2);
  _Float16* Y384  = (_Float16*)alloc((size_t)M_ * FF_ * 2);
  _Float16* QH    = (_Float16*)alloc((size_t)BH_ * S_ * 64 * 2);
  _Float16* KH    = (_Float16*)alloc((size_t)BH_ * S_ * 64 * 2);
  _Float16* VT    = (_Float16*)alloc((size_t)BH_ * S_ * 64 * 2);
  _Float16* PKQ   = (_Float16*)alloc(96 * 96 * 2);
  _Float16* PKK   = (_Float16*)alloc(96 * 96 * 2);
  _Float16* PKV   = (_Float16*)alloc(96 * 96 * 2);
  _Float16* PKO   = (_Float16*)alloc(96 * 96 * 2);
  _Float16* PKI   = (_Float16*)alloc(96 * 384 * 2);
  _Float16* PKW2  = (_Float16*)alloc(384 * 96 * 2);
  _Float16* PKF1  = (_Float16*)alloc(96 * 384 * 2);
  _Float16* PKF2  = (_Float16*)alloc(384 * 96 * 2);
  float*    SC    = (float*)alloc((size_t)B_ * S_ * 4);
  float*    FEAT  = (float*)alloc(B_ * D_ * 4);
  // frontend buffers alias later-phase buffers (strictly sequential use)
  float* CB1 = (float*)QH;                               // [B,48,L]
  float* CB2 = (float*)Y384;                             // [B,96,L]
  float* CB3 = (float*)((char*)Y384 + (((size_t)B_ * D_ * L_ * 4 + 255) & ~(size_t)255));

  dim3 bln(32, 8);
#define GRID1(n) dim3((unsigned)(((n) + 255) / 256))
  auto gemmL = [&](const _Float16* A, const _Float16* Bp, const float* bias,
                   float* o32, _Float16* o16, int K, int N, int act) {
    int tiles = (M_ / 16) * (N / 96);
    gemm_wmma<<<dim3((unsigned)((tiles + 1) / 2)), dim3(32, 2), 0, stream>>>(
        A, Bp, bias, o32, o16, M_, K, N, act);
  };

  // ================= CNN frontend =================
  conv1d_k<<<GRID1(B_ * 48 * L_), 256, 0, stream>>>(x, P(2), P(3), P(4), P(5), CB1, 1, 48, 7, 3, 0, 48, 1, 0);
  conv1d_k<<<GRID1(B_ * 96 * L_), 256, 0, stream>>>(CB1, P(6), P(7), P(8), P(9), CB2, 48, 96, 5, 2, 0, 96, 1, 0);
  conv1d_k<<<GRID1(B_ * 96 * L_), 256, 0, stream>>>(CB2, P(10), P(11), P(12), P(13), CB3, 96, 96, 3, 1, 0, 96, 1, 0);
  conv1d_k<<<GRID1(B_ * 32 * L_), 256, 0, stream>>>(CB3, P(14), P(15), nullptr, nullptr, CB2, 96, 32, 3, 1, 0, 96, 0, 0);
  conv1d_k<<<GRID1(B_ * 32 * L_), 256, 0, stream>>>(CB3, P(16), P(17), nullptr, nullptr, CB2, 96, 32, 7, 3, 32, 96, 0, 0);
  conv1d_k<<<GRID1(B_ * 32 * L_), 256, 0, stream>>>(CB3, P(18), P(19), nullptr, nullptr, CB2, 96, 32, 15, 7, 64, 96, 0, 0);
  bnrelu_k<<<GRID1(B_ * 96 * L_), 256, 0, stream>>>(CB2, P(20), P(21));
  conv1d_k<<<GRID1(B_ * 96 * L_), 256, 0, stream>>>(CB3, P(22), P(23), nullptr, nullptr, CB2, 96, 96, 1, 0, 0, 96, 0, 1);
  ln_pe_pad_k<<<M_ / 8, bln, 0, stream>>>(CB2, P(24), P(25), H32);

  // ================= encoder layers =================
  for (int i = 0; i < NL_; ++i) {
    pack_b<<<GRID1(96 * 96),  256, 0, stream>>>(P(30) + (size_t)i * 96 * 96,  PKQ,  96, 96);
    pack_b<<<GRID1(96 * 96),  256, 0, stream>>>(P(32) + (size_t)i * 96 * 96,  PKK,  96, 96);
    pack_b<<<GRID1(96 * 96),  256, 0, stream>>>(P(34) + (size_t)i * 96 * 96,  PKV,  96, 96);
    pack_b<<<GRID1(96 * 96),  256, 0, stream>>>(P(36) + (size_t)i * 96 * 96,  PKO,  96, 96);
    pack_b<<<GRID1(96 * 384), 256, 0, stream>>>(P(40) + (size_t)i * 96 * 384, PKI,  96, 384);
    pack_b<<<GRID1(384 * 96), 256, 0, stream>>>(P(42) + (size_t)i * 384 * 96, PKW2, 384, 96);
    pack_b<<<GRID1(96 * 384), 256, 0, stream>>>(P(50) + (size_t)i * 96 * 384, PKF1, 96, 384);
    pack_b<<<GRID1(384 * 96), 256, 0, stream>>>(P(52) + (size_t)i * 384 * 96, PKF2, 384, 96);

    ln_emb_k<<<M_ / 8, bln, 0, stream>>>(H32, P(46) + i * D_, P(47) + i * D_,
                                         P(26) + (size_t)i * S_ * D_, P(27) + i * D_,
                                         P(28) + i * D_, P(29) + i * D_, EMB32, X16);
    // Q/K/V projections (WMMA) + head repack (q scaled by 1/sqrt(HD))
    gemmL(X16, PKQ, P(31) + i * D_, Y96, nullptr, 96, 96, 0);
    repack_head_k<<<GRID1(BH_ * S_ * 64), 256, 0, stream>>>(Y96, QH, 0, QSCALE);
    gemmL(X16, PKK, P(33) + i * D_, Y96, nullptr, 96, 96, 0);
    repack_head_k<<<GRID1(BH_ * S_ * 64), 256, 0, stream>>>(Y96, KH, 0, 1.f);
    gemmL(X16, PKV, P(35) + i * D_, Y96, nullptr, 96, 96, 0);
    repack_head_k<<<GRID1(BH_ * S_ * 64), 256, 0, stream>>>(Y96, VT, 1, 1.f);
    // sliding-window attention (WMMA scores + WMMA P*V), writes att f16
    band_attn_k<<<dim3(CH_ * 16, BH_), dim3(32, 8), 0, stream>>>(QH, KH, VT, X16);
    // attn output proj + residual(emb) + LN -> att2 (f32 + f16)
    gemmL(X16, PKO, P(37) + i * D_, Y96, nullptr, 96, 96, 0);
    ln_kernel<<<M_ / 8, bln, 0, stream>>>(Y96, EMB32, P(38) + i * D_, P(39) + i * D_, R32, X16, nullptr);
    // longformer FFN: gelu(att2@iw), @ow + residual(att2) + LN -> h += lf
    gemmL(X16, PKI, P(41) + (size_t)i * FF_, nullptr, Y384, 96, FF_, 2);
    gemmL(Y384, PKW2, P(43) + i * D_, Y96, nullptr, FF_, 96, 0);
    ln_kernel<<<M_ / 8, bln, 0, stream>>>(Y96, R32, P(44) + i * D_, P(45) + i * D_, nullptr, nullptr, H32);
    // custom FFN: h2=LN(h,n2); h += relu(h2@ff1)@ff2
    ln_kernel<<<M_ / 8, bln, 0, stream>>>(H32, nullptr, P(48) + i * D_, P(49) + i * D_, nullptr, X16, nullptr);
    gemmL(X16, PKF1, P(51) + (size_t)i * FF_, nullptr, Y384, 96, FF_, 1);
    gemmL(Y384, PKF2, P(53) + i * D_, Y96, nullptr, FF_, 96, 0);
    add_k<<<GRID1(M_ * D_), 256, 0, stream>>>(H32, Y96, M_ * D_);
  }

  // ================= pooling + head =================
  pool_softmax_k<<<B_, 256, 0, stream>>>(H32, P(54), P(55), SC);
  feat_k<<<B_ * D_, 256, 0, stream>>>(H32, SC, FEAT);
  head_k<<<1, 256, 0, stream>>>(FEAT, P(56), P(57), P(58), P(59), (float*)d_out);
#undef P
#undef GRID1
}
